// hqaf_60687887893234
// MI455X (gfx1250) — matrix operand order repaired and checked by reading, not compiled, not run on hardware
//
#include <hip/hip_runtime.h>
#include <hip/hip_bf16.h>

// ---- CDNA5 vector types ----
typedef __attribute__((ext_vector_type(16))) __bf16 v16bf;
typedef __attribute__((ext_vector_type(8)))  __bf16 v8bf;
typedef __attribute__((ext_vector_type(8)))  float  v8f;
typedef __attribute__((ext_vector_type(4)))  float  v4f;
typedef __attribute__((ext_vector_type(4)))  unsigned int v4u;
typedef __attribute__((ext_vector_type(8)))  int    v8i;
typedef __attribute__((ext_vector_type(4)))  int    v4i;

#define WMMA_BF16(a, b, c) \
  __builtin_amdgcn_wmma_f32_16x16x32_bf16(false, (a), false, (b), (short)0, (c), false, false)

constexpr int Bd = 8, Sd = 1024, Dd = 512, Hd = 8, DKd = 64;

// ---- helpers ----
__device__ __forceinline__ v8bf cvt8(const float* p) {
  v4f x0 = *(const v4f*)p;
  v4f x1 = *(const v4f*)(p + 4);
  v8bf r;
  r[0] = (__bf16)x0[0]; r[1] = (__bf16)x0[1]; r[2] = (__bf16)x0[2]; r[3] = (__bf16)x0[3];
  r[4] = (__bf16)x1[0]; r[5] = (__bf16)x1[1]; r[6] = (__bf16)x1[2]; r[7] = (__bf16)x1[3];
  return r;
}

// 16-bit A(16x32)/B(32x16) fragment per ISA 7.12.2:
//  lanes 0-15 : k = kb .. kb+7 (elems 0..7), kb+16.. (elems 8..15)
//  lanes 16-31: same pattern shifted by 8 (h8=8)
__device__ __forceinline__ v16bf frag_f32(const float* rowptr, int kb, int h8) {
  union { v16bf v; v8bf h[2]; } u;
  u.h[0] = cvt8(rowptr + kb + h8);
  u.h[1] = cvt8(rowptr + kb + 16 + h8);
  return u.v;
}
__device__ __forceinline__ v16bf frag_bf(const __bf16* rowptr, int kb, int h8) {
  union { v16bf v; v8bf h[2]; } u;
  u.h[0] = *(const v8bf*)(rowptr + kb + h8);
  u.h[1] = *(const v8bf*)(rowptr + kb + 16 + h8);
  return u.v;
}

__device__ __forceinline__ float wave_max(float v) {
  #pragma unroll
  for (int o = 16; o > 0; o >>= 1) v = fmaxf(v, __shfl_xor(v, o, 32));
  return v;
}
__device__ __forceinline__ float wave_sum(float v) {
  #pragma unroll
  for (int o = 16; o > 0; o >>= 1) v += __shfl_xor(v, o, 32);
  return v;
}
__device__ __forceinline__ float wave_scan_incl(float v, int lane) {
  #pragma unroll
  for (int o = 1; o < 32; o <<= 1) {
    float t = __shfl_up(v, o, 32);
    if (lane >= o) v += t;
  }
  return v;
}

// ---- Tensor Data Mover: 1-D contiguous tile, global -> LDS (ISA §8 D#) ----
// bytes must be a multiple of 8. Descriptor: data_size=8B, tensor_dim0 =
// tile_dim0 = bytes/8 units, 1-D (tile_dim1 = 0), groups 2/3 unused.
__device__ __forceinline__ void tdm_load_1d(unsigned lds_off,
                                            unsigned long long gaddr,
                                            unsigned bytes) {
  const unsigned units = bytes >> 3;  // 8-byte elements
  v4u g0;
  g0[0] = 1u;                                           // count=1 (valid, user)
  g0[1] = lds_off;                                      // lds_addr[31:0]
  g0[2] = (unsigned)(gaddr & 0xFFFFFFFFull);            // global_addr lo
  g0[3] = (unsigned)((gaddr >> 32) & 0x1FFFFFFull)      // global_addr hi (57b)
          | (2u << 30);                                 // type = 2 ("image")
  v8i g1 = {};
  g1[0] = (int)(3u << 16);                              // data_size = 8 bytes
  g1[1] = (int)((units & 0xFFFFu) << 16);               // tensor_dim0[15:0] @48
  g1[2] = (int)((units >> 16) | (1u << 16));            // dim0 hi | tensor_dim1=1
  g1[3] = (int)((units & 0xFFFFu) << 16);               // tile_dim0 @112
  g1[4] = 0;                                            // tile_dim1/2 = 0 (1-D)
  g1[5] = (int)units;                                   // tensor_dim0_stride lo
  g1[6] = 0;
  g1[7] = 0;
  v4i z = {};
#if __clang_major__ >= 23
  v8i z8 = {};
  __builtin_amdgcn_tensor_load_to_lds(g0, g1, z, z, z8, 0);
#else
  __builtin_amdgcn_tensor_load_to_lds(g0, g1, z, z, 0);
#endif
}

// ============================================================================
// Kernel 1: K = x@Wk^T+bk  (z=0, stored [B,H,S,DK] bf16)
//           V = x@Wv^T+bv  (z=1, stored transposed [B,H,DK,S] bf16)
// ============================================================================
__global__ void __launch_bounds__(32) proj_kv_kernel(
    const float* __restrict__ x,
    const float* __restrict__ Wk, const float* __restrict__ bk,
    const float* __restrict__ Wv, const float* __restrict__ bv,
    __bf16* __restrict__ Kbuf, __bf16* __restrict__ Vbuf) {
  const int lane = threadIdx.x & 31;
  const int hi = lane >> 4, l15 = lane & 15;
  const int which = blockIdx.z;
  const float* W    = which ? Wv : Wk;
  const float* bias = which ? bv : bk;
  const int row0 = blockIdx.x * 16;
  const int col0 = blockIdx.y * 64;

  const float* xr = x + (size_t)(row0 + l15) * Dd;
  v8f acc[4] = {};
  for (int k0 = 0; k0 < Dd; k0 += 32) {
    v16bf a = frag_f32(xr, k0, hi * 8);
    #pragma unroll
    for (int nt = 0; nt < 4; nt++) {
      const float* wr = W + (size_t)(col0 + nt * 16 + l15) * Dd;  // y = x @ W^T
      v16bf bfr = frag_f32(wr, k0, hi * 8);
      acc[nt] = WMMA_BF16(a, bfr, acc[nt]);
    }
  }
  #pragma unroll
  for (int nt = 0; nt < 4; nt++) {
    const int col = col0 + nt * 16 + l15;
    const float bval = bias[col];
    const int h = col >> 6, dk = col & 63;
    #pragma unroll
    for (int v = 0; v < 8; v++) {
      const int row = row0 + v + hi * 8;   // C: VGPR v -> M = v + 8*hi
      const int b = row >> 10, s = row & 1023;
      const float val = acc[nt][v] + bval;
      if (which == 0)
        Kbuf[(((size_t)(b * Hd + h)) * Sd + s) * DKd + dk] = (__bf16)val;
      else
        Vbuf[(((size_t)(b * Hd + h)) * DKd + dk) * Sd + s] = (__bf16)val;
    }
  }
}

// ============================================================================
// Kernel 2: fused causal attention with distance decay.
// One wave per (b,h,16-query tile). K tiles are DMA'd into LDS by the Tensor
// Data Mover (double-buffered on TENSORcnt); the 16xS score stripe lives in
// LDS; both softmaxes + prefix scan + decay + bf16 repack run in place.
// ============================================================================
__global__ void __launch_bounds__(32) attn_kernel(
    const __bf16* __restrict__ Kbuf, const __bf16* __restrict__ Vbuf,
    const float* __restrict__ utT, const float* __restrict__ gammas,
    __bf16* __restrict__ attn_out) {
  __shared__ __bf16 kstage[2][16 * DKd];   // 2 x 2KB TDM staging
  __shared__ float  sc[16][Sd];            // 64 KB score stripe
  const int lane = threadIdx.x & 31;
  const int hi = lane >> 4, l15 = lane & 15;
  const int i0 = blockIdx.x * 16;
  const int h = blockIdx.y, b = blockIdx.z;
  const int bh = b * Hd + h;
  const __bf16* Kb = Kbuf + (size_t)bh * Sd * DKd;
  const __bf16* Vb = Vbuf + (size_t)bh * DKd * Sd;
  const float scale = 0.125f;  // 1/sqrt(64)
  const float g = gammas[h];
  const float gamma = -logf(1.0f + __expf(g));  // -softplus

  // Q fragments (q == k). DK=64 -> two K=32 fragments.
  const __bf16* qr = Kb + (size_t)(i0 + l15) * DKd;
  const v16bf a0 = frag_bf(qr, 0,  hi * 8);
  const v16bf a1 = frag_bf(qr, 32, hi * 8);

  // ---- scores = (Q K^T / 8) * utT, K tiles staged by TDM ----
  const int ntile = (i0 >> 4) + 1;
  const unsigned ks_off0 = (unsigned)(unsigned long long)(const void*)&kstage[0][0];
  const unsigned ks_off1 = (unsigned)(unsigned long long)(const void*)&kstage[1][0];
  const unsigned long long kbase = (unsigned long long)(const void*)Kb;
  const unsigned tile_bytes = 16 * DKd * 2;  // contiguous 2KB per 16 K-rows

  tdm_load_1d(ks_off0, kbase, tile_bytes);                       // prologue
  for (int t = 0; t < ntile; t++) {
    const bool more = (t + 1) < ntile;
    if (more)
      tdm_load_1d((t + 1) & 1 ? ks_off1 : ks_off0,
                  kbase + (unsigned long long)(t + 1) * tile_bytes, tile_bytes);
    if (more) __builtin_amdgcn_s_wait_tensorcnt(1);   // buf t ready
    else      __builtin_amdgcn_s_wait_tensorcnt(0);

    const __bf16* kr = &kstage[t & 1][l15 * DKd];     // ds_load fragments
    v16bf b0 = frag_bf(kr, 0,  hi * 8);
    v16bf b1 = frag_bf(kr, 32, hi * 8);
    v8f c = {};
    c = WMMA_BF16(a0, b0, c);
    c = WMMA_BF16(a1, b1, c);

    const int j0 = t * 16;
    #pragma unroll
    for (int v = 0; v < 8; v++) {
      const int m = v + hi * 8;
      const int i = i0 + m, j = j0 + l15;
      const float u = utT[((size_t)b * Sd + i) * Sd + j];
      sc[m][j] = c[v] * scale * u;
    }
  }
  __syncthreads();

  // ---- per-row: softmax1 -> prefix scan -> decay -> softmax2 -> bf16 P ----
  const int kend = ((i0 + 16) + 31) & ~31;  // P@V K-extent, padded to 32
  for (int r = 0; r < 16; r++) {
    const int i = i0 + r;
    const int L = i + 1;
    float* row = sc[r];

    float mx = -3.0e38f;
    for (int j = lane; j < L; j += 32) mx = fmaxf(mx, row[j]);
    mx = wave_max(mx);

    float sum = 0.f;
    for (int j = lane; j < L; j += 32) sum += __expf(row[j] - mx);
    sum = wave_sum(sum);
    const float inv = 1.0f / sum;

    // s_ = softmax1; suffix = disttotal - distcum = 1 - prefix_incl
    float carry = 0.f, mx2 = -3.0e38f;
    for (int j0 = 0; j0 < L; j0 += 32) {
      const int j = j0 + lane;
      const bool ok = j < L;
      const float raw = ok ? row[j] : 0.f;
      const float s_ = ok ? __expf(raw - mx) * inv : 0.f;
      const float pre = wave_scan_incl(s_, lane) + carry;
      carry = __shfl(pre, 31, 32);
      if (ok) {
        const float Dj = 1.0f - pre;
        const float ds = sqrtf(fmaxf(Dj * fabsf((float)(i - j)), 0.f));
        float te = __expf(ds * gamma);
        te = fminf(fmaxf(te, 1e-5f), 1e5f);
        const float s2 = raw * te;
        row[j] = s2;
        mx2 = fmaxf(mx2, s2);
      }
    }
    mx2 = wave_max(mx2);

    float sum2 = 0.f;
    for (int j = lane; j < L; j += 32) sum2 += __expf(row[j] - mx2);
    sum2 = wave_sum(sum2);
    const float inv2 = 1.0f / sum2;

    // repack P as bf16 in place; zero-pad to kend. Per chunk the wave's LDS
    // loads precede its stores, and bf16 slot j/2 is already consumed.
    __bf16* prow = (__bf16*)row;
    for (int j0 = 0; j0 < kend; j0 += 32) {
      const int j = j0 + lane;
      const float pj = (j < L) ? __expf(row[j] - mx2) * inv2 : 0.f;
      prow[j] = (__bf16)pj;
    }
  }
  __syncthreads();

  // ---- out = P @ V : A from LDS bf16, B = two 16B loads per lane of V^T rows ----
  v8f acc[4] = {};
  for (int jk = 0; jk < kend; jk += 32) {
    if (jk + 32 < kend)
      __builtin_prefetch(Vb + (size_t)l15 * Sd + jk + 32, 0, 0);  // global_prefetch_b8
    const __bf16* pr = (const __bf16*)sc[l15];
    v16bf pa = frag_bf(pr, jk, hi * 8);
    #pragma unroll
    for (int nt = 0; nt < 4; nt++) {
      const int dk = nt * 16 + l15;
      v16bf vbfr = frag_bf(Vb + (size_t)dk * Sd, jk, hi * 8);
      acc[nt] = WMMA_BF16(pa, vbfr, acc[nt]);
    }
  }
  #pragma unroll
  for (int nt = 0; nt < 4; nt++) {
    #pragma unroll
    for (int v = 0; v < 8; v++) {
      const int s = i0 + v + hi * 8;
      const int col = h * DKd + nt * 16 + l15;   // merged-head layout [B,S,D]
      attn_out[((size_t)b * Sd + s) * Dd + col] = (__bf16)acc[nt][v];
    }
  }
}

// ============================================================================
// Kernel 3: out = attn @ Wo^T + bo  (fp32 output)
// ============================================================================
__global__ void __launch_bounds__(32) oproj_kernel(
    const __bf16* __restrict__ A, const float* __restrict__ Wo,
    const float* __restrict__ bo, float* __restrict__ out) {
  const int lane = threadIdx.x & 31;
  const int hi = lane >> 4, l15 = lane & 15;
  const int row0 = blockIdx.x * 16, col0 = blockIdx.y * 64;
  const __bf16* ar = A + (size_t)(row0 + l15) * Dd;
  v8f acc[4] = {};
  for (int k0 = 0; k0 < Dd; k0 += 32) {
    v16bf a = frag_bf(ar, k0, hi * 8);
    #pragma unroll
    for (int nt = 0; nt < 4; nt++) {
      const float* wr = Wo + (size_t)(col0 + nt * 16 + l15) * Dd;
      v16bf bfr = frag_f32(wr, k0, hi * 8);
      acc[nt] = WMMA_BF16(a, bfr, acc[nt]);
    }
  }
  #pragma unroll
  for (int nt = 0; nt < 4; nt++) {
    const int col = col0 + nt * 16 + l15;
    const float bval = bo[col];
    #pragma unroll
    for (int v = 0; v < 8; v++) {
      const int row = row0 + v + hi * 8;
      out[(size_t)row * Dd + col] = acc[nt][v] + bval;
    }
  }
}

// ============================================================================
extern "C" void kernel_launch(void* const* d_in, const int* in_sizes, int n_in,
                              void* d_out, int out_size, void* d_ws, size_t ws_size,
                              hipStream_t stream) {
  const float* x      = (const float*)d_in[0];
  const float* utT    = (const float*)d_in[1];
  const float* Wk     = (const float*)d_in[2];
  const float* bk     = (const float*)d_in[3];
  const float* Wv     = (const float*)d_in[4];
  const float* bv     = (const float*)d_in[5];
  const float* Wo     = (const float*)d_in[6];
  const float* bo     = (const float*)d_in[7];
  const float* gammas = (const float*)d_in[8];

  char* ws = (char*)d_ws;
  __bf16* Kbuf = (__bf16*)(ws);                                   // [B,H,S,DK] 8MB
  __bf16* Vbuf = (__bf16*)(ws + (size_t)8 * 1024 * 1024);         // [B,H,DK,S] 8MB
  __bf16* Abuf = (__bf16*)(ws + (size_t)16 * 1024 * 1024);        // [B,S,D]    8MB
  float* out = (float*)d_out;

  proj_kv_kernel<<<dim3(512, 8, 2), 32, 0, stream>>>(x, Wk, bk, Wv, bv, Kbuf, Vbuf);
  attn_kernel<<<dim3(64, Hd, Bd), 32, 0, stream>>>(Kbuf, Vbuf, utT, gammas, Abuf);
  oproj_kernel<<<dim3(512, 8), 32, 0, stream>>>(Abuf, Wo, bo, out);
}
